// WindowLiteMSA_70050916597972
// MI455X (gfx1250) — matrix operand order repaired and checked by reading, not compiled, not run on hardware
//
#include <hip/hip_runtime.h>
#include <cstdint>

// ---------------------------------------------------------------------------
// WindowLiteMSA fused kernel for MI455X (gfx1250), wave32, WMMA bf16 16x16x32
// ---------------------------------------------------------------------------

typedef __attribute__((ext_vector_type(16))) __bf16 v16bf;
typedef __attribute__((ext_vector_type(8)))  __bf16 v8bf;
typedef __attribute__((ext_vector_type(8)))  float  v8f;
typedef __attribute__((ext_vector_type(4)))  unsigned int v4u;
typedef __attribute__((ext_vector_type(8)))  int v8i;
typedef __attribute__((ext_vector_type(4)))  int v4i;

union ABf { v16bf v; v8bf h[2]; };

constexpr int kB   = 4;
constexpr int kC   = 256;   // input channels
constexpr int kH   = 192;
constexpr int kW   = 192;
constexpr int kWS  = 4;
constexpr int kHn  = kH / kWS;           // 48
constexpr int kWn  = kW / kWS;           // 48
constexpr int kO1  = 768;                // 3*TOT qkv output rows
constexpr int kCW  = 512;                // attention output channels (TOT*2)
constexpr int kO2  = 256;                // proj output channels
constexpr int kPix = kWS * kWS;          // 16
constexpr int kNWin = kB * kHn * kWn;    // 9216

// ---- dynamic LDS layout (regions overlapped across phases) ----
constexpr int MSQ_OFF   = 0;                 // f32 [1536][16] qkv + pooled   (96 KB)
constexpr int X_OFF     = 98304;             // f32 [256][16] raw x window    (16 KB)  phase 1..2
constexpr int ATTNH_OFF = X_OFF;             // bf16 [16][512] attn hi                 phase 5..6
constexpr int BH_OFF    = 114688;            // bf16 [16][256] x hi           ( 8 KB)  phase 2..3
constexpr int BL_OFF    = BH_OFF + 8192;     // bf16 [16][256] x lo           ( 8 KB)  phase 2..3
constexpr int ATTNL_OFF = 114688;            // bf16 [16][512] attn lo                 phase 5..6
constexpr int SMEM_BYTES = 131072;           // 128 KB dynamic LDS

#ifndef __has_builtin
#define __has_builtin(x) 0
#endif
#if __has_builtin(__builtin_amdgcn_tensor_load_to_lds) && __has_builtin(__builtin_amdgcn_s_wait_tensorcnt)
#define HAVE_TDM 1
#else
#define HAVE_TDM 0
#endif

__device__ inline v8f wmma_bf16(v16bf a, v16bf b, v8f c) {
  // D = A(16x32 bf16) * B(32x16 bf16) + C(16x16 f32)
  return __builtin_amdgcn_wmma_f32_16x16x32_bf16(false, a, false, b, (short)0, c, false, false);
}

__device__ inline void split_bf16(float v, unsigned short& hi, unsigned short& lo) {
  __bf16 h = (__bf16)v;
  float hf = (float)h;
  __bf16 l = (__bf16)(v - hf);
  hi = __builtin_bit_cast(unsigned short, h);
  lo = __builtin_bit_cast(unsigned short, l);
}

// ---------------------------------------------------------------------------
// Pre-kernel: split fp32 weights into bf16 hi/lo pairs (for 3x-bf16 WMMA)
// ---------------------------------------------------------------------------
__global__ __launch_bounds__(256) void cvt_w(
    const float* __restrict__ qw, const float* __restrict__ pw,
    unsigned short* __restrict__ qh, unsigned short* __restrict__ ql,
    unsigned short* __restrict__ ph, unsigned short* __restrict__ pl) {
  const int nq = kO1 * kC;    // 196608
  const int np = kO2 * kCW;   // 131072
  int i = blockIdx.x * 256 + threadIdx.x;
  if (i < nq) {
    unsigned short h, l; split_bf16(qw[i], h, l); qh[i] = h; ql[i] = l;
  } else if (i < nq + np) {
    int j = i - nq;
    unsigned short h, l; split_bf16(pw[j], h, l); ph[j] = h; pl[j] = l;
  }
}

// ---------------------------------------------------------------------------
// Fused: TDM stage -> qkv WMMA GEMM -> pool -> linear attention -> proj WMMA
// One block per 4x4 window, 256 threads (8 wave32).
// ---------------------------------------------------------------------------
__global__ __launch_bounds__(256) void fused_wlmsa(
    const float* __restrict__ x,
    const unsigned short* __restrict__ qh, const unsigned short* __restrict__ ql,
    const unsigned short* __restrict__ ph, const unsigned short* __restrict__ pl,
    const float* __restrict__ bn_g, const float* __restrict__ bn_b,
    const float* __restrict__ bn_m, const float* __restrict__ bn_v,
    float* __restrict__ out)
{
  extern __shared__ char smem[];
  float* msq = (float*)(smem + MSQ_OFF);   // [1536][16]
  float* xw  = (float*)(smem + X_OFF);     // [256][16]

  const int tid = threadIdx.x;
  const int lane = tid & 31;
  const int wv   = tid >> 5;      // wave id 0..7
  const int hl   = lane >> 4;     // half-wave 0/1
  const int m15  = lane & 15;

  const int wid = blockIdx.x;
  const int b   = wid / (kHn * kWn);
  const int rem = wid % (kHn * kWn);
  const int h0  = (rem / kWn) * kWS;
  const int w0  = (rem % kWn) * kWS;

  // ---------------- Phase 1: stage x window [256ch][16px] into LDS ----------
#if HAVE_TDM
  if (wv == 0) {
    unsigned lds_dst = (unsigned)(uintptr_t)(smem + X_OFF);
    unsigned long long ga =
        (unsigned long long)(uintptr_t)(x + (((size_t)b * kC) * kH + h0) * kW + w0);
    // D# group0: count=1, lds_addr, global_addr, type=2
    v4u g0 = { 1u, lds_dst, (unsigned)ga,
               (unsigned)((ga >> 32) & 0x1FFFFFFull) | (2u << 30) };
    // D# group1: data_size=4B; tensor_dim0=W, tensor_dim1=H;
    // tile = (4, 4, 256); dim0_stride=W, dim1_stride=H*W
    v8i g1 = { (int)(2u << 16),
               (int)(((unsigned)kW & 0xFFFFu) << 16),
               (int)((((unsigned)kW) >> 16) | (((unsigned)kH & 0xFFFFu) << 16)),
               (int)((((unsigned)kH) >> 16) | (4u << 16)),
               (int)(4u | (256u << 16)),
               (int)(kW),
               (int)((((unsigned)(kH * kW)) & 0xFFFFu) << 16),
               (int)(((unsigned)(kH * kW)) >> 16) };
    // D# group2: tensor_dim2=C, dim2_stride=H*W, tile_dim3=0 (3-D tile)
    v4i g2 = { (int)kC, 0, (int)(kH * kW), 0 };
    v4i g3 = { 0, 0, 0, 0 };
#if __clang_major__ >= 23
    v8i gz = {0,0,0,0,0,0,0,0};
    __builtin_amdgcn_tensor_load_to_lds(g0, g1, g2, g3, gz, 0);
#else
    __builtin_amdgcn_tensor_load_to_lds(g0, g1, g2, g3, 0);
#endif
    __builtin_amdgcn_s_wait_tensorcnt(0);
  }
#else
  for (int i = tid; i < kC * kPix; i += 256) {
    int c = i >> 4, p = i & 15;
    xw[c * 16 + p] =
        x[(((size_t)b * kC + c) * kH + (h0 + (p >> 2))) * kW + (w0 + (p & 3))];
  }
#endif
  __syncthreads();

  // -------- Phase 2: build transposed bf16 hi/lo B operand [px][256] -------
  {
    unsigned short* bh = (unsigned short*)(smem + BH_OFF);
    unsigned short* bl = (unsigned short*)(smem + BL_OFF);
    for (int i = tid; i < kC * kPix; i += 256) {
      int p = i >> 8;         // pixel 0..15
      int c = i & 255;        // channel 0..255
      unsigned short h, l; split_bf16(xw[c * 16 + p], h, l);
      bh[p * 256 + c] = h; bl[p * 256 + c] = l;
    }
  }
  __syncthreads();

  // -------- Phase 3: qkv GEMM (768x256)x(256x16), 3x-bf16 WMMA -------------
  for (int t = 0; t < 6; ++t) {                 // 48 M-tiles / 8 waves
    const int o0 = (wv * 6 + t) * 16;
    const int row = o0 + m15;
    v8f acc = {0.f,0.f,0.f,0.f,0.f,0.f,0.f,0.f};
    for (int k0 = 0; k0 < kC; k0 += 32) {
      ABf ahi, alo;
      const size_t ab = (size_t)row * kC + k0 + 16 * hl;  // A: K=16*half+e
      ahi.h[0] = *(const v8bf*)(qh + ab);
      ahi.h[1] = *(const v8bf*)(qh + ab + 8);
      alo.h[0] = *(const v8bf*)(ql + ab);
      alo.h[1] = *(const v8bf*)(ql + ab + 8);
      const int boff = m15 * 512 + (k0 + 16 * hl) * 2;    // B: K=16*half+e
      v16bf bhi = *(const v16bf*)(smem + BH_OFF + boff);
      v16bf blo = *(const v16bf*)(smem + BL_OFF + boff);
      acc = wmma_bf16(ahi.v, bhi, acc);
      acc = wmma_bf16(ahi.v, blo, acc);
      acc = wmma_bf16(alo.v, bhi, acc);
    }
#pragma unroll
    for (int r = 0; r < 8; ++r)                 // C layout: M=r+8*half, N=lane&15
      msq[(o0 + r + 8 * hl) * 16 + m15] = acc[r];
  }
  __syncthreads();

  // -------- Phase 4: 5x5 avg-pool (pad 2, fixed /25) inside window ---------
  for (int ch = tid; ch < kO1; ch += 256) {
    const float* r = msq + ch * 16;
    float v[16];
#pragma unroll
    for (int i = 0; i < 16; ++i) v[i] = r[i];
    float cf[4], ta[4], tb[4];
#pragma unroll
    for (int j = 0; j < 4; ++j) {
      cf[j] = v[j] + v[4 + j] + v[8 + j] + v[12 + j];  // rows 0..3
      ta[j] = cf[j] - v[12 + j];                       // rows {0,1,2}
      tb[j] = cf[j] - v[j];                            // rows {1,2,3}
    }
    float* o = msq + (kO1 + ch) * 16;
#pragma unroll
    for (int i = 0; i < 4; ++i) {
      float t0 = (i == 0) ? ta[0] : (i == 3) ? tb[0] : cf[0];
      float t1 = (i == 0) ? ta[1] : (i == 3) ? tb[1] : cf[1];
      float t2 = (i == 0) ? ta[2] : (i == 3) ? tb[2] : cf[2];
      float t3 = (i == 0) ? ta[3] : (i == 3) ? tb[3] : cf[3];
      float rf = t0 + t1 + t2 + t3;
      o[i * 4 + 0] = (rf - t3) * 0.04f;
      o[i * 4 + 1] = rf * 0.04f;
      o[i * 4 + 2] = rf * 0.04f;
      o[i * 4 + 3] = (rf - t0) * 0.04f;
    }
  }
  __syncthreads();

  // -------- Phase 5: linear attention, 64 groups x 16 px -------------------
  {
    const int g = tid >> 2;            // group 0..63 (>=32 -> pooled half)
    const int quad = tid & 3;          // 4 pixels per thread
    const int cb = (g < 32) ? (24 * g) : (kO1 + 24 * (g - 32));
    float kv[8][9];
#pragma unroll
    for (int d = 0; d < 8; ++d)
#pragma unroll
      for (int m = 0; m < 9; ++m) kv[d][m] = 0.f;
    for (int l = 0; l < 16; ++l) {
      float kl[8], vl[8];
#pragma unroll
      for (int d = 0; d < 8; ++d) kl[d] = fmaxf(msq[(cb + 8 + d) * 16 + l], 0.f);
#pragma unroll
      for (int m = 0; m < 8; ++m) vl[m] = msq[(cb + 16 + m) * 16 + l];
#pragma unroll
      for (int d = 0; d < 8; ++d) {
#pragma unroll
        for (int m = 0; m < 8; ++m) kv[d][m] += kl[d] * vl[m];
        kv[d][8] += kl[d];             // v padded with ones column
      }
    }
    unsigned short* ah = (unsigned short*)(smem + ATTNH_OFF);
    unsigned short* al = (unsigned short*)(smem + ATTNL_OFF);
    for (int li = 0; li < 4; ++li) {
      const int l = quad * 4 + li;
      float q[8];
#pragma unroll
      for (int d = 0; d < 8; ++d) q[d] = fmaxf(msq[(cb + d) * 16 + l], 0.f);
      float num[9];
#pragma unroll
      for (int m = 0; m < 9; ++m) {
        float s = 0.f;
#pragma unroll
        for (int d = 0; d < 8; ++d) s += q[d] * kv[d][m];
        num[m] = s;
      }
      const float dnm = 1.f / (num[8] + 1e-15f);
#pragma unroll
      for (int m = 0; m < 8; ++m) {
        unsigned short h, lo;
        split_bf16(num[m] * dnm, h, lo);
        ah[l * kCW + (g * 8 + m)] = h;
        al[l * kCW + (g * 8 + m)] = lo;
      }
    }
  }
  __syncthreads();

  // -------- Phase 6: proj GEMM (256x512)x(512x16) + BN + store -------------
  for (int t = 0; t < 2; ++t) {                 // 16 M-tiles / 8 waves
    const int o0 = (wv * 2 + t) * 16;
    const int row = o0 + m15;
    v8f acc = {0.f,0.f,0.f,0.f,0.f,0.f,0.f,0.f};
    for (int k0 = 0; k0 < kCW; k0 += 32) {
      ABf ahi, alo;
      const size_t ab = (size_t)row * kCW + k0 + 16 * hl;
      ahi.h[0] = *(const v8bf*)(ph + ab);
      ahi.h[1] = *(const v8bf*)(ph + ab + 8);
      alo.h[0] = *(const v8bf*)(pl + ab);
      alo.h[1] = *(const v8bf*)(pl + ab + 8);
      const int boff = m15 * (kCW * 2) + (k0 + 16 * hl) * 2;
      v16bf bhi = *(const v16bf*)(smem + ATTNH_OFF + boff);
      v16bf blo = *(const v16bf*)(smem + ATTNL_OFF + boff);
      acc = wmma_bf16(ahi.v, bhi, acc);
      acc = wmma_bf16(ahi.v, blo, acc);
      acc = wmma_bf16(alo.v, bhi, acc);
    }
    const int l = m15, hh = h0 + (l >> 2), ww = w0 + (l & 3);
#pragma unroll
    for (int r = 0; r < 8; ++r) {
      const int o = o0 + r + 8 * hl;
      const float inv = bn_g[o] * rsqrtf(bn_v[o] + 1e-5f);
      const float y = acc[r] * inv + (bn_b[o] - bn_m[o] * inv);
      out[(((size_t)b * kO2 + o) * kH + hh) * kW + ww] = y;
    }
  }
}

// ---------------------------------------------------------------------------
extern "C" void kernel_launch(void* const* d_in, const int* in_sizes, int n_in,
                              void* d_out, int out_size, void* d_ws, size_t ws_size,
                              hipStream_t stream) {
  (void)in_sizes; (void)n_in; (void)out_size; (void)ws_size;
  const float* x      = (const float*)d_in[0];
  const float* qkv_w  = (const float*)d_in[1];
  const float* proj_w = (const float*)d_in[2];
  const float* bn_g   = (const float*)d_in[3];
  const float* bn_b   = (const float*)d_in[4];
  const float* bn_m   = (const float*)d_in[5];
  const float* bn_v   = (const float*)d_in[6];

  unsigned short* qh = (unsigned short*)d_ws;          // 768*256 bf16 hi
  unsigned short* ql = qh + kO1 * kC;                  // 768*256 bf16 lo
  unsigned short* ph = ql + kO1 * kC;                  // 256*512 bf16 hi
  unsigned short* pl = ph + kO2 * kCW;                 // 256*512 bf16 lo

  (void)hipFuncSetAttribute((const void*)fused_wlmsa,
                            hipFuncAttributeMaxDynamicSharedMemorySize, SMEM_BYTES);

  const int ncvt = kO1 * kC + kO2 * kCW;               // 327680
  cvt_w<<<dim3((ncvt + 255) / 256), dim3(256), 0, stream>>>(
      qkv_w, proj_w, qh, ql, ph, pl);

  fused_wlmsa<<<dim3(kNWin), dim3(256), SMEM_BYTES, stream>>>(
      x, qh, ql, ph, pl, bn_g, bn_b, bn_m, bn_v, (float*)d_out);
}
